// PointNetSetAbstractionMsg_1726576857082
// MI455X (gfx1250) — matrix operand order repaired and checked by reading, not compiled, not run on hardware
//
#include <hip/hip_runtime.h>
#include <hip/hip_bf16.h>

typedef __attribute__((ext_vector_type(16))) _Float16 v16h;
typedef __attribute__((ext_vector_type(8)))  _Float16 v8h;
typedef __attribute__((ext_vector_type(8)))  float    v8f;

#define B_ 16
#define N_ 4096
#define S_ 1024
#define OUTC_ 320

// ---------------- FPS: one block per batch, xyz + dist cached in LDS ----------------
__global__ __launch_bounds__(1024) void fps_kernel(const float* __restrict__ xyz,
                                                   int* __restrict__ fps_idx) {
    __shared__ float sx[N_], sy[N_], sz[N_], sdist[N_];
    __shared__ float wmax[32];
    __shared__ int   warg[32];
    __shared__ int   bcast;
    const int b = blockIdx.x, t = threadIdx.x;
    const float* xb = xyz + (size_t)b * 3 * N_;
    for (int j = t; j < N_; j += 1024) {
        sx[j] = xb[j]; sy[j] = xb[N_ + j]; sz[j] = xb[2 * N_ + j];
        sdist[j] = 1e10f;
    }
    __syncthreads();
    int far = 0;
    const int w = t >> 5, lane = t & 31;
    for (int i = 0; i < S_; ++i) {
        if (t == 0) fps_idx[b * S_ + i] = far;
        const float cx = sx[far], cy = sy[far], cz = sz[far];
        float bv = -1.0f; int bi = 0;
        for (int j = t; j < N_; j += 1024) {
            float dx = sx[j] - cx, dy = sy[j] - cy, dz = sz[j] - cz;
            float d = dx * dx + dy * dy + dz * dz;
            float od = sdist[j];
            d = d < od ? d : od;
            sdist[j] = d;
            if (d > bv || (d == bv && j < bi)) { bv = d; bi = j; }
        }
        #pragma unroll
        for (int off = 16; off; off >>= 1) {
            float ov = __shfl_xor(bv, off);
            int   oi = __shfl_xor(bi, off);
            if (ov > bv || (ov == bv && oi < bi)) { bv = ov; bi = oi; }
        }
        if (lane == 0) { wmax[w] = bv; warg[w] = bi; }
        __syncthreads();
        if (w == 0) {
            bv = wmax[lane]; bi = warg[lane];
            #pragma unroll
            for (int off = 16; off; off >>= 1) {
                float ov = __shfl_xor(bv, off);
                int   oi = __shfl_xor(bi, off);
                if (ov > bv || (ov == bv && oi < bi)) { bv = ov; bi = oi; }
            }
            if (lane == 0) bcast = bi;
        }
        __syncthreads();
        far = bcast;
    }
}

// gather new_xyz into ws (B,S,3) and output (B,3,S)
__global__ void newxyz_kernel(const float* __restrict__ xyz, const int* __restrict__ fps_idx,
                              float* __restrict__ nxyz, float* __restrict__ out) {
    int i = blockIdx.x * 256 + threadIdx.x;
    if (i >= B_ * S_) return;
    int b = i >> 10, s = i & 1023;
    int idx = fps_idx[i];
    const float* xb = xyz + (size_t)b * 3 * N_;
    float x = xb[idx], y = xb[N_ + idx], z = xb[2 * N_ + idx];
    nxyz[i * 3 + 0] = x; nxyz[i * 3 + 1] = y; nxyz[i * 3 + 2] = z;
    out[(size_t)b * 3 * S_ + 0 * S_ + s] = x;
    out[(size_t)b * 3 * S_ + 1 * S_ + s] = y;
    out[(size_t)b * 3 * S_ + 2 * S_ + s] = z;
}

// ---------------- Ball query: one wave per query, ordered compaction ----------------
__global__ __launch_bounds__(256) void query_ball_kernel(const float* __restrict__ xyz,
                                                         const float* __restrict__ nxyz,
                                                         int* __restrict__ gidx,
                                                         float r2, int K) {
    int wid = (blockIdx.x * 256 + threadIdx.x) >> 5;
    int lane = threadIdx.x & 31;
    if (wid >= B_ * S_) return;               // wave-uniform exit
    int b = wid >> 10;
    const float* xb = xyz + (size_t)b * 3 * N_;
    float qx = nxyz[wid * 3 + 0], qy = nxyz[wid * 3 + 1], qz = nxyz[wid * 3 + 2];
    int* out = gidx + (size_t)wid * K;
    int cnt = 0, first = -1;
    for (int base = 0; base < N_ && cnt < K; base += 32) {
        int n = base + lane;
        float dx = xb[n] - qx, dy = xb[N_ + n] - qy, dz = xb[2 * N_ + n] - qz;
        bool inr = (dx * dx + dy * dy + dz * dz) <= r2;
        unsigned mask = (unsigned)__ballot(inr);
        if (first < 0 && mask) first = base + __ffs(mask) - 1;
        if (inr) {
            int pos = cnt + __popc(mask & ((1u << lane) - 1u));
            if (pos < K) out[pos] = n;
        }
        cnt += __popc(mask);
    }
    if (cnt < K) {
        if (cnt == 0) first = 0;
        for (int j = cnt + lane; j < K; j += 32) out[j] = first;
    }
}

// ---------------- Build layer-0 input rows (f16, stride 32, channels 6..31 = 0) ----------------
__global__ void build_kernel(const float* __restrict__ points, const float* __restrict__ xyz,
                             const float* __restrict__ nxyz, const int* __restrict__ gidx,
                             _Float16* __restrict__ X0, int total, int kshift) {
    int row = blockIdx.x * 256 + threadIdx.x;
    if (row >= total) return;
    int q = row >> kshift;          // b*S + s
    int b = q >> 10;
    int idx = gidx[row];
    const float* pb = points + (size_t)b * 3 * N_;
    const float* xb = xyz + (size_t)b * 3 * N_;
    _Float16* o = X0 + (size_t)row * 32;
    o[0] = (_Float16)pb[idx];
    o[1] = (_Float16)pb[N_ + idx];
    o[2] = (_Float16)pb[2 * N_ + idx];
    o[3] = (_Float16)(xb[idx]          - nxyz[q * 3 + 0]);
    o[4] = (_Float16)(xb[N_ + idx]     - nxyz[q * 3 + 1]);
    o[5] = (_Float16)(xb[2 * N_ + idx] - nxyz[q * 3 + 2]);
    #pragma unroll
    for (int j = 6; j < 32; ++j) o[j] = (_Float16)0.0f;
}

// ---------------- Pack weights into the B-fragment layout (contiguous per lane) ----------------
// Wp[(kt*Cout + n)*32 + kk] = W[n][kt*32+kk]  (zero-padded past Cin)
__global__ void packw_kernel(const float* __restrict__ W, _Float16* __restrict__ Wp,
                             int Cout, int Cin, int CinPad) {
    int i = blockIdx.x * 256 + threadIdx.x;
    if (i >= Cout * CinPad) return;
    int n = i / CinPad, k = i - n * CinPad;
    float v = (k < Cin) ? W[n * Cin + k] : 0.0f;
    Wp[((size_t)((k >> 5) * Cout + n)) * 32 + (k & 31)] = (_Float16)v;
}

__global__ void zero_kernel(float* __restrict__ p, int n) {
    int i = blockIdx.x * 256 + threadIdx.x;
    if (i < n) p[i] = 0.0f;
}

// ---------------- WMMA GEMM: Y = X * W^T + b, f16 in / f32 accumulate / f16 out ----------------
// NKT = Cin/32 and COUT are compile-time so B-fragments live in fixed VGPRs, all loops
// unroll, and every stride is a constant (no v_mul_u64 in the hot loop).
// MT=2 independent row tiles per step for load/WMMA overlap.
template <int NKT, int COUT>
__global__ __launch_bounds__(256) void gemm_kernel(const _Float16* __restrict__ X,
                                                   const _Float16* __restrict__ Wp,
                                                   const float* __restrict__ bias,
                                                   _Float16* __restrict__ Y,
                                                   float* __restrict__ stats,
                                                   int rowTiles) {
    const int Cin = NKT * 32;
    __shared__ float lsum[16], lsq[16];
    const int t = threadIdx.x, w = t >> 5, lane = t & 31;
    const int n = lane & 15, half = lane >> 4;
    const int colBase = blockIdx.y * 16;
    if (t < 16) { lsum[t] = 0.0f; lsq[t] = 0.0f; }
    __syncthreads();

    v16h bf[NKT];
    #pragma unroll
    for (int kt = 0; kt < NKT; ++kt)
        bf[kt] = *(const v16h*)(Wp + ((size_t)(kt * COUT + colBase + n)) * 32 + half * 16);
    const float bn = bias[colBase + n];

    float s = 0.0f, s2 = 0.0f;
    const int ITER = 4, MT = 2;                        // 8 row tiles per wave
    for (int it = 0; it < ITER; ++it) {
        int rt0 = (blockIdx.x * 8 + w) * (ITER * MT) + it * MT;   // wave-uniform
        if (rt0 >= rowTiles) break;                    // rowTiles always even
        const _Float16* xr0 = X + (size_t)(rt0 * 16 + n) * Cin;
        const _Float16* xr1 = xr0 + (size_t)16 * Cin;

        v8h alo0[NKT], ahi0[NKT], alo1[NKT], ahi1[NKT];
        #pragma unroll
        for (int kt = 0; kt < NKT; ++kt) {
            alo0[kt] = *(const v8h*)(xr0 + kt * 32 + half * 16);
            ahi0[kt] = *(const v8h*)(xr0 + kt * 32 + half * 16 + 8);
            alo1[kt] = *(const v8h*)(xr1 + kt * 32 + half * 16);
            ahi1[kt] = *(const v8h*)(xr1 + kt * 32 + half * 16 + 8);
        }
        v8f acc0 = {}, acc1 = {};
        #pragma unroll
        for (int kt = 0; kt < NKT; ++kt) {
            v16h a0 = __builtin_shufflevector(alo0[kt], ahi0[kt], 0, 1, 2, 3, 4, 5, 6, 7,
                                              8, 9, 10, 11, 12, 13, 14, 15);
            acc0 = __builtin_amdgcn_wmma_f32_16x16x32_f16(
                false, a0, false, bf[kt], (short)0, acc0, false, false);
            v16h a1 = __builtin_shufflevector(alo1[kt], ahi1[kt], 0, 1, 2, 3, 4, 5, 6, 7,
                                              8, 9, 10, 11, 12, 13, 14, 15);
            acc1 = __builtin_amdgcn_wmma_f32_16x16x32_f16(
                false, a1, false, bf[kt], (short)0, acc1, false, false);
        }
        _Float16* yr0 = Y + (size_t)rt0 * 16 * COUT + (size_t)(half * 8) * COUT + colBase + n;
        _Float16* yr1 = yr0 + (size_t)16 * COUT;
        #pragma unroll
        for (int r = 0; r < 8; ++r) {                  // D elem: m = half*8+r, col = colBase+n
            float v0 = acc0[r] + bn;
            yr0[(size_t)r * COUT] = (_Float16)v0;      // constant immediate offsets
            s += v0; s2 += v0 * v0;
            float v1 = acc1[r] + bn;
            yr1[(size_t)r * COUT] = (_Float16)v1;
            s += v1; s2 += v1 * v1;
        }
    }
    s  += __shfl_xor(s, 16);
    s2 += __shfl_xor(s2, 16);
    if (lane < 16) { atomicAdd(&lsum[n], s); atomicAdd(&lsq[n], s2); }
    __syncthreads();
    if (t < 16) {
        atomicAdd(&stats[colBase + t], lsum[t]);
        atomicAdd(&stats[COUT + colBase + t], lsq[t]);
    }
}

// per-channel: a = gamma*rsqrt(var+eps), c = beta - mean*a
__global__ void finalize_kernel(const float* __restrict__ stats, const float* __restrict__ gamma,
                                const float* __restrict__ beta, float* __restrict__ ab,
                                int C, float invCount) {
    int c = blockIdx.x * 128 + threadIdx.x;
    if (c >= C) return;
    float mean = stats[c] * invCount;
    float var  = stats[C + c] * invCount - mean * mean;
    float a = gamma[c] * rsqrtf(var + 1e-5f);
    ab[c] = a;
    ab[C + c] = beta[c] - mean * a;
}

// in-place y = relu(a*y + c), vectorized 8 halves (16B) per lane
__global__ void bnrelu_kernel(_Float16* __restrict__ Y, const float* __restrict__ ab,
                              int C, size_t total) {
    size_t i = ((size_t)blockIdx.x * 256 + threadIdx.x) * 8;
    if (i >= total) return;
    int c0 = (int)(i % (size_t)C);                    // 8 | C, so c0..c0+7 stay in-row
    v8h y = *(const v8h*)(Y + i);
    v8h o;
    #pragma unroll
    for (int j = 0; j < 8; ++j) {
        float v = (float)y[j] * ab[c0 + j] + ab[C + c0 + j];
        o[j] = (_Float16)(v > 0.0f ? v : 0.0f);
    }
    *(v8h*)(Y + i) = o;
}

// max over K, write fp32 output slice (B, 320, S)
__global__ void maxpool_kernel(const _Float16* __restrict__ Y, float* __restrict__ out,
                               int K, int C, int chOff) {
    int i = blockIdx.x * 256 + threadIdx.x;
    if (i >= B_ * S_ * C) return;
    int c = i % C, q = i / C;
    int b = q >> 10, s = q & 1023;
    const _Float16* yq = Y + ((size_t)q * K) * C + c;
    float m = -1e30f;
    for (int k = 0; k < K; ++k) {
        float v = (float)yq[(size_t)k * C];
        m = v > m ? v : m;
    }
    out[(size_t)B_ * 3 * S_ + ((size_t)b * OUTC_ + chOff + c) * S_ + s] = m;
}

extern "C" void kernel_launch(void* const* d_in, const int* in_sizes, int n_in,
                              void* d_out, int out_size, void* d_ws, size_t ws_size,
                              hipStream_t stream) {
    (void)in_sizes; (void)n_in; (void)out_size; (void)ws_size;
    const float* xyz    = (const float*)d_in[0];
    const float* points = (const float*)d_in[1];
    auto P = [&](int s, int l, int which) -> const float* {
        return (const float*)d_in[2 + (s * 3 + l) * 4 + which];
    };

    // workspace carve
    char* ws = (char*)d_ws;
    size_t off = 0;
    auto carve = [&](size_t bytes) -> void* {
        void* p = ws + off;
        off = (off + bytes + 255) & ~(size_t)255;
        return p;
    };
    const size_t ROWS_MAX = (size_t)B_ * S_ * 128;
    int*       fps_idx = (int*)carve((size_t)B_ * S_ * 4);
    float*     nxyz    = (float*)carve((size_t)B_ * S_ * 3 * 4);
    int*       gidx    = (int*)carve(ROWS_MAX * 4);
    float*     stats   = (float*)carve(2 * 128 * 4);
    float*     ab      = (float*)carve(2 * 128 * 4);
    _Float16*  Wpack   = (_Float16*)carve((size_t)128 * 128 * 2);
    _Float16*  ping    = (_Float16*)carve(ROWS_MAX * 128 * 2);
    _Float16*  pong    = (_Float16*)carve(ROWS_MAX * 128 * 2);   // X0 overlays pong

    float* out = (float*)d_out;

    fps_kernel<<<B_, 1024, 0, stream>>>(xyz, fps_idx);
    newxyz_kernel<<<(B_ * S_ + 255) / 256, 256, 0, stream>>>(xyz, fps_idx, nxyz, out);

    const int   Ks[3]      = {16, 32, 128};
    const int   kshift[3]  = {4, 5, 7};
    const float radii[3]   = {0.1f, 0.2f, 0.4f};
    const int   mlps[3][3] = {{32, 32, 64}, {64, 64, 128}, {64, 96, 128}};
    const int   cins[3][3] = {{6, 32, 32}, {6, 64, 64}, {6, 64, 96}};
    const int   chOffs[3]  = {0, 64, 192};

    for (int sc = 0; sc < 3; ++sc) {
        const int K = Ks[sc];
        const int rows = B_ * S_ * K;
        const float r2 = radii[sc] * radii[sc];

        query_ball_kernel<<<(B_ * S_) / 8, 256, 0, stream>>>(xyz, nxyz, gidx, r2, K);
        build_kernel<<<(rows + 255) / 256, 256, 0, stream>>>(points, xyz, nxyz, gidx,
                                                             pong, rows, kshift[sc]);
        const _Float16* in = pong;
        for (int l = 0; l < 3; ++l) {
            const int CinA   = cins[sc][l];
            const int CinPad = (l == 0) ? 32 : CinA;
            const int Cout   = mlps[sc][l];
            _Float16* outBuf = (l == 1) ? pong : ping;

            packw_kernel<<<(Cout * CinPad + 255) / 256, 256, 0, stream>>>(
                P(sc, l, 0), Wpack, Cout, CinA, CinPad);
            zero_kernel<<<1, 256, 0, stream>>>(stats, 2 * Cout);

            const int rowTiles = rows / 16;
            const int gx = (rowTiles + 63) / 64;   // 8 waves * (ITER*MT=8) row-tiles each
            dim3 grid(gx, Cout / 16);
            const int cfg = ((CinPad >> 5) << 8) | Cout;
            switch (cfg) {
                case (1 << 8) | 32:
                    gemm_kernel<1, 32><<<grid, 256, 0, stream>>>(
                        in, Wpack, P(sc, l, 1), outBuf, stats, rowTiles); break;
                case (1 << 8) | 64:
                    gemm_kernel<1, 64><<<grid, 256, 0, stream>>>(
                        in, Wpack, P(sc, l, 1), outBuf, stats, rowTiles); break;
                case (2 << 8) | 64:
                    gemm_kernel<2, 64><<<grid, 256, 0, stream>>>(
                        in, Wpack, P(sc, l, 1), outBuf, stats, rowTiles); break;
                case (2 << 8) | 96:
                    gemm_kernel<2, 96><<<grid, 256, 0, stream>>>(
                        in, Wpack, P(sc, l, 1), outBuf, stats, rowTiles); break;
                case (2 << 8) | 128:
                    gemm_kernel<2, 128><<<grid, 256, 0, stream>>>(
                        in, Wpack, P(sc, l, 1), outBuf, stats, rowTiles); break;
                default:
                    gemm_kernel<3, 128><<<grid, 256, 0, stream>>>(
                        in, Wpack, P(sc, l, 1), outBuf, stats, rowTiles); break;
            }

            finalize_kernel<<<(Cout + 127) / 128, 128, 0, stream>>>(
                stats, P(sc, l, 2), P(sc, l, 3), ab, Cout, 1.0f / (float)rows);

            const size_t total = (size_t)rows * Cout;
            bnrelu_kernel<<<(unsigned)((total / 8 + 255) / 256), 256, 0, stream>>>(
                outBuf, ab, Cout, total);
            in = outBuf;
        }
        const int Cl = mlps[sc][2];
        maxpool_kernel<<<(B_ * S_ * Cl + 255) / 256, 256, 0, stream>>>(
            ping, out, K, Cl, chOffs[sc]);
    }
}
